// CARP_Decoder_4389456577018
// MI455X (gfx1250) — compile-verified, matrix operand order checked by reading, not run on hardware
//
#include <hip/hip_runtime.h>
#include <hip/hip_bf16.h>

// CARP decoder step for MI455X (gfx1250): bf16 WMMA (fp32 accumulate) for all
// GEMM stages; per-lane contiguous b128 operand loads (weights pre-transposed
// to bf16 col-major, V stored transposed [qd][n]); flash-style online softmax
// with 32-wide N chunks and full-K P@V WMMA. bf16 packing via v_perm_b32
// (1 instr per pair after +0x8000 rounding); scalar stores use d16_hi folding.

constexpr int B  = 64;
constexpr int P  = 256;
constexpr int N  = 1024;
constexpr int E  = 128;
constexpr int QD = 16;
constexpr int HD = 128;           // H*QD
constexpr float INV_SQRT_E = 0.08838834764831845f;  // 1/sqrt(128)

typedef __attribute__((ext_vector_type(16))) __bf16 v16bf;
typedef __attribute__((ext_vector_type(8)))  float  v8f;

// ---- packed fp32 -> 2x bf16 (low = a, high = b) --------------------------
__device__ __forceinline__ unsigned int pk2bf(float a, float b) {
#if __has_builtin(__builtin_amdgcn_cvt_pk_bf16_f32)
    auto r = __builtin_amdgcn_cvt_pk_bf16_f32(a, b);
    return __builtin_bit_cast(unsigned int, r);
#else
    // round-half-up then take high halves; one v_perm_b32 packs both.
    const unsigned int ua = __float_as_uint(a) + 0x8000u;
    const unsigned int ub = __float_as_uint(b) + 0x8000u;
    return __builtin_amdgcn_perm(ub, ua, 0x07060302u);
#endif
}
// scalar: shift folds into global_store_d16_hi_b16
__device__ __forceinline__ unsigned short f2bu(float f) {
    return (unsigned short)((__float_as_uint(f) + 0x8000u) >> 16);
}
__device__ __forceinline__ float fast_tanh(float x) {
#if __has_builtin(__builtin_amdgcn_tanhf)
    return __builtin_amdgcn_tanhf(x);
#else
    const float t = __expf(2.0f * x);
    return (t - 1.0f) / (t + 1.0f);
#endif
}

__device__ __forceinline__ v8f wmma_bf16(v16bf a, v16bf b, v8f c) {
    return __builtin_amdgcn_wmma_f32_16x16x32_bf16(false, a, false, b,
                                                   (short)0, c, false, false);
}

// WMMA operand viewed as 8 packed dwords / two b128 quads -- no lane moves.
union BfVec {
    unsigned int u[8];
    uint4        q[2];
    v16bf        v;
};

// Two b128 loads -> full 16x bf16 operand.
__device__ __forceinline__ v16bf ld16bf(const unsigned short* p0,
                                        const unsigned short* p1) {
    BfVec c;
    c.q[0] = *reinterpret_cast<const uint4*>(p0);
    c.q[1] = *reinterpret_cast<const uint4*>(p1);
    return c.v;
}
// One b128 load into the low half, upper K half zero (for qd=16 padding).
__device__ __forceinline__ v16bf ld8bf_lo(const unsigned short* p) {
    BfVec c;
    c.q[0] = *reinterpret_cast<const uint4*>(p);
    c.q[1] = uint4{0u, 0u, 0u, 0u};
    return c.v;
}
// 8 consecutive fp32 -> 8 packed bf16 (four dwords).
__device__ __forceinline__ uint4 cvt8(const float* p) {
    const float4 f0 = *reinterpret_cast<const float4*>(p);
    const float4 f1 = *reinterpret_cast<const float4*>(p + 4);
    uint4 r;
    r.x = pk2bf(f0.x, f0.y);
    r.y = pk2bf(f0.z, f0.w);
    r.z = pk2bf(f1.x, f1.y);
    r.w = pk2bf(f1.z, f1.w);
    return r;
}

// ---------------------------------------------------------------- stage 0
// Transpose + convert the four 128x128 weight blocks to bf16 col-major.
__global__ void prep_weights(const float* __restrict__ Wq,
                             const float* __restrict__ Wk,
                             const float* __restrict__ Wv,
                             const float* __restrict__ Wc,
                             unsigned short* __restrict__ WqT,
                             unsigned short* __restrict__ WkT,
                             unsigned short* __restrict__ WvT,
                             unsigned short* __restrict__ WcT) {
    const int idx = blockIdx.x * blockDim.x + threadIdx.x;   // 4*16384 threads
    const int m   = idx >> 14;
    const int r   = idx & 16383;
    const int col = r >> 7;
    const int e   = r & 127;
    const float* src = (m == 0) ? Wq : (m == 1) ? Wk : (m == 2) ? Wv : Wc;
    unsigned short* dst = (m == 0) ? WqT : (m == 1) ? WkT : (m == 2) ? WvT : WcT;
    dst[col * 128 + e] = f2bu(src[(size_t)e * 128 + col]);
}

// ---------------------------------------------------------------- stage 1
// K = enc @ Wk (row-major bf16), Vt = (enc @ Wv)^T ([b][qd][n] bf16),
// plus bf16 copy of enc_nodes.
__global__ void kv_proj(const float* __restrict__ enc,
                        const unsigned short* __restrict__ WkT,
                        const unsigned short* __restrict__ WvT,
                        unsigned short* __restrict__ Kb,
                        unsigned short* __restrict__ Vt,
                        unsigned short* __restrict__ ENb) {
    const int b     = blockIdx.x >> 6;
    const int ntile = blockIdx.x & 63;
    const int wave  = threadIdx.x >> 5;
    const int lane  = threadIdx.x & 31;
    const int m16   = lane & 15;
    const int kbase = (lane >> 4) << 3;   // 0 or 8
    const int n0    = ntile * 16;
    const int col   = wave * 16 + m16;

    const float* encRow   = enc + ((size_t)b * N + n0 + m16) * E;
    unsigned short* enRow = ENb + ((size_t)b * N + n0 + m16) * E;
    const unsigned short* wkc = WkT + (size_t)col * 128;
    const unsigned short* wvc = WvT + (size_t)col * 128;

    v8f ck = {}, cv = {};
    #pragma unroll
    for (int ec = 0; ec < 4; ++ec) {
        const int e0 = ec * 32 + kbase;
        BfVec A;
        A.q[0] = cvt8(encRow + e0);
        A.q[1] = cvt8(encRow + e0 + 16);
        if (wave == 0) {
            *reinterpret_cast<uint4*>(enRow + e0)      = A.q[0];
            *reinterpret_cast<uint4*>(enRow + e0 + 16) = A.q[1];
        }
        const v16bf bk = ld16bf(wkc + e0, wkc + e0 + 16);
        const v16bf bv = ld16bf(wvc + e0, wvc + e0 + 16);
        ck = wmma_bf16(A.v, bk, ck);
        cv = wmma_bf16(A.v, bv, cv);
    }
    // K row-major: coalesced across lanes (d16_hi b16 stores)
    #pragma unroll
    for (int r = 0; r < 8; ++r)
        Kb[((size_t)b * N + n0 + r + kbase) * HD + col] = f2bu(ck[r]);
    // V transposed: a lane's 8 output rows are consecutive n -> one b128 store
    {
        uint4 q;
        q.x = pk2bf(cv[0], cv[1]);
        q.y = pk2bf(cv[2], cv[3]);
        q.z = pk2bf(cv[4], cv[5]);
        q.w = pk2bf(cv[6], cv[7]);
        *reinterpret_cast<uint4*>(Vt + ((size_t)b * HD + col) * N + n0 + kbase) = q;
    }
}

// ---------------------------------------------------------------- stage 2
// Q = [enc_last | load] @ Wq  (last contraction row as rank-1 fixup)
__global__ void q_proj(const float* __restrict__ encLast,
                       const float* __restrict__ loadv,
                       const float* __restrict__ Wq,
                       const unsigned short* __restrict__ WqT,
                       unsigned short* __restrict__ Qb) {
    const int b     = blockIdx.x >> 4;
    const int ptile = blockIdx.x & 15;
    const int wave  = threadIdx.x >> 5;
    const int lane  = threadIdx.x & 31;
    const int m16   = lane & 15;
    const int kbase = (lane >> 4) << 3;
    const int p0    = ptile * 16;
    const int col   = wave * 16 + m16;

    const float* arow = encLast + ((size_t)b * P + p0 + m16) * E;
    const unsigned short* wqc = WqT + (size_t)col * 128;

    v8f c = {};
    #pragma unroll
    for (int ec = 0; ec < 4; ++ec) {
        const int e0 = ec * 32 + kbase;
        BfVec A;
        A.q[0] = cvt8(arow + e0);
        A.q[1] = cvt8(arow + e0 + 16);
        const v16bf bb = ld16bf(wqc + e0, wqc + e0 + 16);
        c = wmma_bf16(A.v, bb, c);
    }
    const float wlast = Wq[(size_t)E * HD + col];   // fp32 row 128 of Wq
    #pragma unroll
    for (int r = 0; r < 8; ++r) {
        const int p = p0 + r + kbase;
        Qb[((size_t)b * P + p) * HD + col] = f2bu(c[r] + loadv[b * P + p] * wlast);
    }
}

// ---------------------------------------------------------------- stage 3
// Flash attention per (b, h, p-tile), N in chunks of 32. S^T = K_chunk @ Q^T
// puts the score tiles directly in the A-operand layout of the full-K P@V.
__global__ void attn(const unsigned short* __restrict__ Qb,
                     const unsigned short* __restrict__ Kb,
                     const unsigned short* __restrict__ Vt,
                     const float* __restrict__ mask,
                     unsigned short* __restrict__ AttOut) {
    const int b     = blockIdx.x >> 4;
    const int ptile = blockIdx.x & 15;
    const int h     = threadIdx.x >> 5;   // 8 waves = 8 heads
    const int lane  = threadIdx.x & 31;
    const int m16   = lane & 15;
    const int kbase = (lane >> 4) << 3;
    const int p0    = ptile * 16;

    // B operand of S^T; upper K half zero (qd padded 16->32)
    const v16bf qb = ld8bf_lo(Qb + ((size_t)b * P + p0 + m16) * HD + h * QD + kbase);

    const float* maskRow = mask + ((size_t)b * P + p0 + m16) * N;
    const unsigned short* kptr  = Kb + ((size_t)b * N + m16) * HD + h * QD + kbase;
    const unsigned short* vtptr = Vt + ((size_t)b * HD + h * QD + m16) * N;

    float mrun = -3.0e38f, lrun = 0.0f;
    v8f o = {};

    for (int nc = 0; nc < 32; ++nc) {
        const int n0 = nc * 32;
        if (nc < 31) {   // global_prefetch_b8 of next chunk
            __builtin_prefetch(kptr + (size_t)(n0 + 32) * HD, 0, 1);
            __builtin_prefetch(vtptr + n0 + 32, 0, 1);
        }
        // two 16-row score tiles: A = K rows, K-dim = qd (padded)
        const v16bf ka0 = ld8bf_lo(kptr + (size_t)n0 * HD);
        const v16bf ka1 = ld8bf_lo(kptr + (size_t)(n0 + 16) * HD);
        v8f s0 = {}, s1 = {};
        s0 = wmma_bf16(ka0, qb, s0);
        s1 = wmma_bf16(ka1, qb, s1);

        // scores: sv[i] = score(p0+m16, ...) for this lane's 16 n values
        const float4 mA = *reinterpret_cast<const float4*>(maskRow + n0 + kbase);
        const float4 mB = *reinterpret_cast<const float4*>(maskRow + n0 + kbase + 4);
        const float4 mC = *reinterpret_cast<const float4*>(maskRow + n0 + 16 + kbase);
        const float4 mD = *reinterpret_cast<const float4*>(maskRow + n0 + 20 + kbase);
        float sv[16];
        sv[0]  = s0[0] * 0.25f + mA.x;  sv[1]  = s0[1] * 0.25f + mA.y;
        sv[2]  = s0[2] * 0.25f + mA.z;  sv[3]  = s0[3] * 0.25f + mA.w;
        sv[4]  = s0[4] * 0.25f + mB.x;  sv[5]  = s0[5] * 0.25f + mB.y;
        sv[6]  = s0[6] * 0.25f + mB.z;  sv[7]  = s0[7] * 0.25f + mB.w;
        sv[8]  = s1[0] * 0.25f + mC.x;  sv[9]  = s1[1] * 0.25f + mC.y;
        sv[10] = s1[2] * 0.25f + mC.z;  sv[11] = s1[3] * 0.25f + mC.w;
        sv[12] = s1[4] * 0.25f + mD.x;  sv[13] = s1[5] * 0.25f + mD.y;
        sv[14] = s1[6] * 0.25f + mD.z;  sv[15] = s1[7] * 0.25f + mD.w;

        float lm = sv[0];
        #pragma unroll
        for (int i = 1; i < 16; ++i) lm = fmaxf(lm, sv[i]);
        lm = fmaxf(lm, __shfl_xor(lm, 16));
        const float mnew  = fmaxf(mrun, lm);
        const float alpha = __expf(mrun - mnew);

        float ex[16];
        float ladd = 0.0f;
        #pragma unroll
        for (int i = 0; i < 16; ++i) {
            ex[i] = __expf(sv[i] - mnew);
            ladd += ex[i];
        }
        ladd += __shfl_xor(ladd, 16);
        lrun = lrun * alpha + ladd;
        mrun = mnew;

        BfVec PA;                      // full K=32 A operand of P@V
        #pragma unroll
        for (int i = 0; i < 8; ++i) PA.u[i] = pk2bf(ex[2 * i], ex[2 * i + 1]);

        // B = V chunk from Vt: per-lane contiguous along n
        const v16bf vbv = ld16bf(vtptr + n0 + kbase, vtptr + n0 + 16 + kbase);

        #pragma unroll
        for (int r = 0; r < 8; ++r)
            o[r] *= __shfl(alpha, r + kbase);
        o = wmma_bf16(PA.v, vbv, o);
    }

    #pragma unroll
    for (int r = 0; r < 8; ++r) {
        const float lp = __shfl(lrun, r + kbase);
        const int p = p0 + r + kbase;
        AttOut[((size_t)b * P + p) * HD + h * QD + m16] = f2bu(o[r] / lp);
    }
}

// ---------------------------------------------------------------- stage 4
// MH = AttOut @ Wc_w + Wc_b
__global__ void mh_proj(const unsigned short* __restrict__ AttOut,
                        const unsigned short* __restrict__ WcT,
                        const float* __restrict__ Wc_b,
                        unsigned short* __restrict__ MH) {
    const int b     = blockIdx.x >> 4;
    const int ptile = blockIdx.x & 15;
    const int wave  = threadIdx.x >> 5;
    const int lane  = threadIdx.x & 31;
    const int m16   = lane & 15;
    const int kbase = (lane >> 4) << 3;
    const int p0    = ptile * 16;
    const int col   = wave * 16 + m16;

    const unsigned short* arow = AttOut + ((size_t)b * P + p0 + m16) * HD;
    const unsigned short* wcc  = WcT + (size_t)col * 128;

    v8f c = {};
    #pragma unroll
    for (int ec = 0; ec < 4; ++ec) {
        const int e0 = ec * 32 + kbase;
        const v16bf a  = ld16bf(arow + e0, arow + e0 + 16);
        const v16bf bb = ld16bf(wcc + e0,  wcc + e0 + 16);
        c = wmma_bf16(a, bb, c);
    }
    const float bias = Wc_b[col];
    #pragma unroll
    for (int r = 0; r < 8; ++r) {
        const int p = p0 + r + kbase;
        MH[((size_t)b * P + p) * HD + col] = f2bu(c[r] + bias);
    }
}

// ---------------------------------------------------------------- stage 5
// sh_score = MH @ enc^T, 10*tanh(s/sqrt(E)) + mask, softmax over N.
__global__ void final_softmax(const unsigned short* __restrict__ MH,
                              const unsigned short* __restrict__ ENb,
                              const float* __restrict__ mask,
                              float* __restrict__ out) {
    __shared__ float redmax[8 * 16];
    __shared__ float redsum[8 * 16];

    const int b     = blockIdx.x >> 4;
    const int ptile = blockIdx.x & 15;
    const int wave  = threadIdx.x >> 5;   // each wave owns 8 n-tiles
    const int lane  = threadIdx.x & 31;
    const int m16   = lane & 15;
    const int kbase = (lane >> 4) << 3;
    const int p0    = ptile * 16;

    v16bf aA[4];
    {
        const unsigned short* mrow = MH + ((size_t)b * P + p0 + m16) * HD;
        #pragma unroll
        for (int ec = 0; ec < 4; ++ec)
            aA[ec] = ld16bf(mrow + ec * 32 + kbase, mrow + ec * 32 + 16 + kbase);
    }

    float val[8][8];
    float rmax[8];
    #pragma unroll
    for (int r = 0; r < 8; ++r) rmax[r] = -3.0e38f;

    #pragma unroll
    for (int t = 0; t < 8; ++t) {
        const int n0 = (wave * 8 + t) * 16;
        const unsigned short* erow = ENb + ((size_t)b * N + n0 + m16) * E;
        v8f c = {};
        #pragma unroll
        for (int ec = 0; ec < 4; ++ec) {
            const v16bf bb = ld16bf(erow + ec * 32 + kbase,
                                    erow + ec * 32 + 16 + kbase);
            c = wmma_bf16(aA[ec], bb, c);
        }
        #pragma unroll
        for (int r = 0; r < 8; ++r) {
            const int p = p0 + r + kbase;
            const float v = 10.0f * fast_tanh(c[r] * INV_SQRT_E)
                          + mask[((size_t)b * P + p) * N + n0 + m16];
            val[t][r] = v;
            rmax[r] = fmaxf(rmax[r], v);
        }
    }
    #pragma unroll
    for (int r = 0; r < 8; ++r) {
        float m = rmax[r];
        m = fmaxf(m, __shfl_xor(m, 1));
        m = fmaxf(m, __shfl_xor(m, 2));
        m = fmaxf(m, __shfl_xor(m, 4));
        m = fmaxf(m, __shfl_xor(m, 8));
        rmax[r] = m;
    }
    if (m16 == 0) {
        #pragma unroll
        for (int r = 0; r < 8; ++r) redmax[wave * 16 + r + kbase] = rmax[r];
    }
    __syncthreads();

    float gmax[8], gsum[8];
    #pragma unroll
    for (int r = 0; r < 8; ++r) {
        float m = -3.0e38f;
        #pragma unroll
        for (int w = 0; w < 8; ++w) m = fmaxf(m, redmax[w * 16 + r + kbase]);
        gmax[r] = m;
        gsum[r] = 0.0f;
    }
    #pragma unroll
    for (int t = 0; t < 8; ++t)
        #pragma unroll
        for (int r = 0; r < 8; ++r) {
            const float e = __expf(val[t][r] - gmax[r]);
            val[t][r] = e;
            gsum[r] += e;
        }
    #pragma unroll
    for (int r = 0; r < 8; ++r) {
        float s = gsum[r];
        s += __shfl_xor(s, 1);
        s += __shfl_xor(s, 2);
        s += __shfl_xor(s, 4);
        s += __shfl_xor(s, 8);
        gsum[r] = s;
    }
    if (m16 == 0) {
        #pragma unroll
        for (int r = 0; r < 8; ++r) redsum[wave * 16 + r + kbase] = gsum[r];
    }
    __syncthreads();

    #pragma unroll
    for (int r = 0; r < 8; ++r) {
        float s = 0.0f;
        #pragma unroll
        for (int w = 0; w < 8; ++w) s += redsum[w * 16 + r + kbase];
        const float inv = 1.0f / s;
        const int p = p0 + r + kbase;
        #pragma unroll
        for (int t = 0; t < 8; ++t) {
            const int n = (wave * 8 + t) * 16 + m16;
            out[((size_t)b * P + p) * N + n] = val[t][r] * inv;  // coalesced across lanes
        }
    }
}

// ---------------------------------------------------------------- launcher
extern "C" void kernel_launch(void* const* d_in, const int* in_sizes, int n_in,
                              void* d_out, int out_size, void* d_ws, size_t ws_size,
                              hipStream_t stream) {
    (void)in_sizes; (void)n_in; (void)out_size; (void)ws_size;

    const float* encLast = (const float*)d_in[0];
    const float* loadv   = (const float*)d_in[1];
    const float* mask    = (const float*)d_in[2];
    const float* enc     = (const float*)d_in[3];
    const float* Wq      = (const float*)d_in[4];
    const float* Wk      = (const float*)d_in[5];
    const float* Wv      = (const float*)d_in[6];
    const float* Wc_w    = (const float*)d_in[7];
    const float* Wc_b    = (const float*)d_in[8];
    float* out = (float*)d_out;

    char* ws = (char*)d_ws;
    size_t off = 0;
    auto take = [&](size_t bytes) {
        char* p = ws + off;
        off += (bytes + 255) & ~(size_t)255;
        return p;
    };
    unsigned short* Qb     = (unsigned short*)take((size_t)B * P * HD * 2);
    unsigned short* Kb     = (unsigned short*)take((size_t)B * N * HD * 2);
    unsigned short* Vt     = (unsigned short*)take((size_t)B * HD * N * 2);
    unsigned short* ENb    = (unsigned short*)take((size_t)B * N * E  * 2);
    unsigned short* AttOut = (unsigned short*)take((size_t)B * P * HD * 2);
    unsigned short* MHb    = (unsigned short*)take((size_t)B * P * HD * 2);
    unsigned short* WqT    = (unsigned short*)take((size_t)128 * 128 * 2);
    unsigned short* WkT    = (unsigned short*)take((size_t)128 * 128 * 2);
    unsigned short* WvT    = (unsigned short*)take((size_t)128 * 128 * 2);
    unsigned short* WcT    = (unsigned short*)take((size_t)128 * 128 * 2);

    const dim3 blk(256);
    prep_weights <<<dim3(256),    blk, 0, stream>>>(Wq, Wk, Wv, Wc_w, WqT, WkT, WvT, WcT);
    kv_proj      <<<dim3(B * 64), blk, 0, stream>>>(enc, WkT, WvT, Kb, Vt, ENb);
    q_proj       <<<dim3(B * 16), blk, 0, stream>>>(encLast, loadv, Wq, WqT, Qb);
    attn         <<<dim3(B * 16), blk, 0, stream>>>(Qb, Kb, Vt, mask, AttOut);
    mh_proj      <<<dim3(B * 16), blk, 0, stream>>>(AttOut, WcT, Wc_b, MHb);
    final_softmax<<<dim3(B * 16), blk, 0, stream>>>(MHb, ENb, mask, out);
}
